// MaskedDecoder_88699664597180
// MI455X (gfx1250) — matrix operand order repaired and checked by reading, not compiled, not run on hardware
//
#include <hip/hip_runtime.h>

typedef _Float16 half_t;
typedef __attribute__((ext_vector_type(16))) _Float16 v16h;
typedef __attribute__((ext_vector_type(8)))  float    v8f;

namespace {
constexpr int Bb  = 4;
constexpr int Tt  = 1024;
constexpr int Dd  = 1024;
constexpr int Hh  = 16;
constexpr int HDd = 64;
constexpr int DFFc = 4096;
constexpr int LLc  = 4;
constexpr int MM  = Bb * Tt;   // 4096 rows
}

// Build a v16h fragment from two 16-byte chunks.
static __device__ __forceinline__ v16h make_frag(uint4 lo, uint4 hi) {
  union { uint4 u[2]; v16h h; } r;
  r.u[0] = lo; r.u[1] = hi;
  return r.h;
}

// A-matrix 16x32 f16 fragment: lanes 0-15 hold K=[0..15],
// lanes 16-31 hold K=[16..31]; row M = lane&15 (rowptr pre-offset).
static __device__ __forceinline__ v16h load_a_frag(const half_t* rowptr, int k0, int lanehi) {
  const half_t* p = rowptr + k0 + (lanehi ? 16 : 0);
  return make_frag(*(const uint4*)p, *(const uint4*)(p + 8));
}

// B fragment from weights stored in fragment-native layout [(K/16)][N][16].
static __device__ __forceinline__ v16h load_b_frag(const half_t* Wf, int kh, int n, int N) {
  const half_t* p = Wf + ((size_t)kh * N + n) * 16;
  return make_frag(*(const uint4*)p, *(const uint4*)(p + 8));
}

// One k-step worth of fragments for a 32x64 wave tile.
struct Frags { v16h a0, a1, b0, b1, b2, b3; };

static __device__ __forceinline__ Frags
load_frags(const half_t* ar0, const half_t* ar1, const half_t* Wf,
           int k0, int n0, int lanehi, int lanelo, int N) {
  Frags f;
  f.a0 = load_a_frag(ar0, k0, lanehi);
  f.a1 = load_a_frag(ar1, k0, lanehi);
  int kh = (k0 >> 4) + lanehi;
  f.b0 = load_b_frag(Wf, kh, n0 + lanelo, N);
  f.b1 = load_b_frag(Wf, kh, n0 + 16 + lanelo, N);
  f.b2 = load_b_frag(Wf, kh, n0 + 32 + lanelo, N);
  f.b3 = load_b_frag(Wf, kh, n0 + 48 + lanelo, N);
  return f;
}

static __device__ __forceinline__ void mma8(const Frags& f, v8f acc[2][4]) {
  acc[0][0] = __builtin_amdgcn_wmma_f32_16x16x32_f16(false, f.a0, false, f.b0, (short)0, acc[0][0], false, false);
  acc[1][0] = __builtin_amdgcn_wmma_f32_16x16x32_f16(false, f.a1, false, f.b0, (short)0, acc[1][0], false, false);
  acc[0][1] = __builtin_amdgcn_wmma_f32_16x16x32_f16(false, f.a0, false, f.b1, (short)0, acc[0][1], false, false);
  acc[1][1] = __builtin_amdgcn_wmma_f32_16x16x32_f16(false, f.a1, false, f.b1, (short)0, acc[1][1], false, false);
  acc[0][2] = __builtin_amdgcn_wmma_f32_16x16x32_f16(false, f.a0, false, f.b2, (short)0, acc[0][2], false, false);
  acc[1][2] = __builtin_amdgcn_wmma_f32_16x16x32_f16(false, f.a1, false, f.b2, (short)0, acc[1][2], false, false);
  acc[0][3] = __builtin_amdgcn_wmma_f32_16x16x32_f16(false, f.a0, false, f.b3, (short)0, acc[0][3], false, false);
  acc[1][3] = __builtin_amdgcn_wmma_f32_16x16x32_f16(false, f.a1, false, f.b3, (short)0, acc[1][3], false, false);
}

// ---------------------------------------------------------------------------
// Weight repack: fp32 [K,N] row-major -> f16 B-fragment layout [(K/16)][N][16]
// ---------------------------------------------------------------------------
__global__ void __launch_bounds__(256)
wconv_kernel(const float* __restrict__ src, half_t* __restrict__ dst, int K, int N) {
  size_t idx = (size_t)blockIdx.x * 256 + threadIdx.x;   // grid sized to K*N/256
  int k = (int)(idx / (size_t)N);
  int n = (int)(idx % (size_t)N);
  dst[((size_t)(k >> 4) * N + n) * 16 + (k & 15)] = (half_t)src[idx];
}

// ---------------------------------------------------------------------------
// Embedding + sinusoidal positional encoding. x fp32 + f16 shadow copy.
// ---------------------------------------------------------------------------
__global__ void __launch_bounds__(256)
embed_kernel(const int* __restrict__ tok, const float* __restrict__ emb,
             float* __restrict__ x, half_t* __restrict__ xh) {
  int bt = blockIdx.x;
  int t  = bt & (Tt - 1);
  int tk = tok[bt];
  const float* er = emb + (size_t)tk * Dd;
#pragma unroll
  for (int j = 0; j < 4; ++j) {
    int d = threadIdx.x + j * 256;
    float ang = (float)t * expf((float)(d & ~1) * (-9.2103403719761836f / (float)Dd));
    float pe  = (d & 1) ? cosf(ang) : sinf(ang);
    float val = er[d] + pe;
    x[(size_t)bt * Dd + d]  = val;
    xh[(size_t)bt * Dd + d] = (half_t)val;
  }
}

// ---------------------------------------------------------------------------
// f16 WMMA GEMM:  C[M,N] = A[M,K] (f16) * Wfrag[K,N] (f16) + bias
// 128 threads = 4 waves; block tile 128x64; wave tile 32x64; K step 32.
// Double-buffered with 2x k-unroll so buffers alternate roles (no reg copies,
// no WMMA WAR-hazard NOPs from rotation movs).  Requires K % 64 == 0.
// ---------------------------------------------------------------------------
__global__ void __launch_bounds__(128)
gemm_f16(const half_t* __restrict__ A, const half_t* __restrict__ Wf,
         const float* __restrict__ bias, float* __restrict__ Cf,
         half_t* __restrict__ Ch, int N, int K, int relu) {
  int lane   = threadIdx.x & 31;
  int wave   = threadIdx.x >> 5;
  int lanehi = lane >> 4;
  int lanelo = lane & 15;
  int row0 = blockIdx.y * 128 + wave * 32;
  int n0   = blockIdx.x * 64;

  const half_t* ar0 = A + (size_t)(row0 + lanelo) * K;
  const half_t* ar1 = ar0 + (size_t)16 * K;

  v8f acc[2][4] = {};

  Frags f0 = load_frags(ar0, ar1, Wf, 0, n0, lanehi, lanelo, N);
  int k0 = 0;
  for (; k0 + 64 < K; k0 += 64) {
    Frags f1 = load_frags(ar0, ar1, Wf, k0 + 32, n0, lanehi, lanelo, N);
    mma8(f0, acc);
    f0 = load_frags(ar0, ar1, Wf, k0 + 64, n0, lanehi, lanelo, N);
    mma8(f1, acc);
  }
  {
    Frags f1 = load_frags(ar0, ar1, Wf, k0 + 32, n0, lanehi, lanelo, N);
    mma8(f0, acc);
    mma8(f1, acc);
  }

  // Epilogue: bias / ReLU, fp32 and/or f16 stores.
#pragma unroll
  for (int grp = 0; grp < 2; ++grp) {
    int mbase = row0 + grp * 16 + lanehi * 8;
#pragma unroll
    for (int t = 0; t < 4; ++t) {
      int col = n0 + t * 16 + lanelo;
      float bv = bias ? bias[col] : 0.0f;
#pragma unroll
      for (int r = 0; r < 8; ++r) {
        float val = acc[grp][t][r] + bv;
        if (relu) val = fmaxf(val, 0.0f);
        size_t off = (size_t)(mbase + r) * N + col;
        if (Cf) Cf[off] = val;
        if (Ch) Ch[off] = (half_t)val;
      }
    }
  }
}

// ---------------------------------------------------------------------------
// Causal flash attention. Grid: (T/64, H, B). Block: 128 (4 waves).
// Wave owns 16 q rows; iterate keys in steps of 32 with online softmax.
// ---------------------------------------------------------------------------
__global__ void __launch_bounds__(128)
attn_flash(const half_t* __restrict__ Q, const half_t* __restrict__ Km,
           const half_t* __restrict__ V, half_t* __restrict__ O) {
  __shared__ __align__(16) half_t ldsV[4 * 32 * 16];  // V tile in B-frag layout
  __shared__ __align__(16) half_t ldsP[4 * 16 * 32];  // per-wave P scratch

  int tid    = threadIdx.x;
  int lane   = tid & 31;
  int wave   = tid >> 5;
  int lanehi = lane >> 4;
  int lanelo = lane & 15;
  int b  = blockIdx.z;
  int h  = blockIdx.y;
  int q0 = blockIdx.x * 64 + wave * 16;

  size_t bbase = (size_t)b * Tt * Dd + (size_t)h * HDd;
  const half_t* qrow = Q + bbase + (size_t)(q0 + lanelo) * Dd;
  v16h qa0 = load_a_frag(qrow, 0, lanehi);
  v16h qa1 = load_a_frag(qrow, 32, lanehi);

  v8f  o[4] = {};
  float mx[8], sm[8];
#pragma unroll
  for (int r = 0; r < 8; ++r) { mx[r] = -1e30f; sm[r] = 0.0f; }
  const float scale = 0.125f;  // 1/sqrt(64)

  int nkt = blockIdx.x * 2 + 2;  // keys up to blockIdx.x*64 + 63 (uniform per WG)
  for (int kt = 0; kt < nkt; ++kt) {
    int kbase = kt * 32;

    // ---- S = Q * K^T for 32 keys (direct global K fragments) ----
    float sc[2][8];
#pragma unroll
    for (int g = 0; g < 2; ++g) {
      const half_t* kp = Km + bbase + (size_t)(kbase + g * 16 + lanelo) * Dd + lanehi * 16;
      v16h kb0 = make_frag(*(const uint4*)kp,        *(const uint4*)(kp + 8));
      v16h kb1 = make_frag(*(const uint4*)(kp + 32), *(const uint4*)(kp + 40));
      v8f s = {};
      s = __builtin_amdgcn_wmma_f32_16x16x32_f16(false, qa0, false, kb0, (short)0, s, false, false);
      s = __builtin_amdgcn_wmma_f32_16x16x32_f16(false, qa1, false, kb1, (short)0, s, false, false);
#pragma unroll
      for (int r = 0; r < 8; ++r) {
        int qg = q0 + lanehi * 8 + r;
        int kg = kbase + g * 16 + lanelo;
        sc[g][r] = (kg <= qg) ? s[r] * scale : -1e30f;
      }
    }

    // ---- online softmax (row reductions across 16-lane halves) ----
#pragma unroll
    for (int r = 0; r < 8; ++r) {
      float tmax = fmaxf(sc[0][r], sc[1][r]);
#pragma unroll
      for (int d = 1; d < 16; d <<= 1) tmax = fmaxf(tmax, __shfl_xor(tmax, d, 32));
      float newm = fmaxf(mx[r], tmax);
      float corr = __expf(mx[r] - newm);
      float p0 = __expf(sc[0][r] - newm);
      float p1 = __expf(sc[1][r] - newm);
      float rs = p0 + p1;
#pragma unroll
      for (int d = 1; d < 16; d <<= 1) rs += __shfl_xor(rs, d, 32);
      sm[r] = sm[r] * corr + rs;
      mx[r] = newm;
      sc[0][r] = p0; sc[1][r] = p1;
#pragma unroll
      for (int tile = 0; tile < 4; ++tile) o[tile][r] *= corr;
    }

    // ---- stage V tile into LDS (B-frag layout) + P into per-wave LDS ----
#pragma unroll
    for (int i = 0; i < 2; ++i) {
      int flat = tid * 8 + i * 1024;        // 2048 halves total
      int kk = flat >> 6;                   // 0..31 key within tile
      int n  = flat & 63;                   // hd col, multiple of 8
      const half_t* vp = V + bbase + (size_t)(kbase + kk) * Dd + n;
      uint4 dword = *(const uint4*)vp;
      const half_t* hp = (const half_t*)&dword;
#pragma unroll
      for (int j = 0; j < 8; ++j) {
        int n8 = n + j;
        int tile = n8 >> 4;
        int l = ((kk >> 4) << 4) | (n8 & 15);
        ldsV[((tile * 32 + l) << 4) | (kk & 15)] = hp[j];
      }
    }
    half_t* pp = &ldsP[wave * 512];
#pragma unroll
    for (int g = 0; g < 2; ++g)
#pragma unroll
      for (int r = 0; r < 8; ++r) {
        int m = lanehi * 8 + r;
        pp[m * 32 + g * 16 + lanelo] = (half_t)sc[g][r];
      }
    __syncthreads();

    // ---- O += P * V ----
    v16h pa = load_a_frag(pp + lanelo * 32, 0, lanehi);
#pragma unroll
    for (int tile = 0; tile < 4; ++tile) {
      const half_t* vb = &ldsV[(tile * 32 + lane) << 4];
      v16h bf = make_frag(*(const uint4*)vb, *(const uint4*)(vb + 8));
      o[tile] = __builtin_amdgcn_wmma_f32_16x16x32_f16(
          false, pa, false, bf, (short)0, o[tile], false, false);
    }
    __syncthreads();  // protect LDS before next iteration's stores
  }

  // ---- normalize and write out (f16) ----
#pragma unroll
  for (int tile = 0; tile < 4; ++tile) {
#pragma unroll
    for (int r = 0; r < 8; ++r) {
      int m = lanehi * 8 + r;
      int col = h * HDd + tile * 16 + lanelo;
      float val = o[tile][r] / sm[r];
      O[(size_t)(b * Tt + q0 + m) * Dd + col] = (half_t)val;
    }
  }
}

// ---------------------------------------------------------------------------
// Fused residual add + LayerNorm (fp32), emits fp32 and f16 copies.
// One block (256 threads) per row of D=1024.
// ---------------------------------------------------------------------------
__global__ void __launch_bounds__(256)
ln_res(const float* __restrict__ xin, const float* __restrict__ y,
       const float* __restrict__ g, const float* __restrict__ bt,
       float* __restrict__ xout, half_t* __restrict__ xh) {
  __shared__ float sb[8];
  int row = blockIdx.x;
  int tid = threadIdx.x;
  const float* xr = xin + (size_t)row * Dd;
  const float* yr = y   + (size_t)row * Dd;

  float v[4];
  float s = 0.0f;
#pragma unroll
  for (int j = 0; j < 4; ++j) {
    int d = tid + j * 256;
    v[j] = xr[d] + yr[d];
    s += v[j];
  }
#pragma unroll
  for (int d = 16; d >= 1; d >>= 1) s += __shfl_xor(s, d, 32);
  if ((tid & 31) == 0) sb[tid >> 5] = s;
  __syncthreads();
  float tot = 0.0f;
#pragma unroll
  for (int i = 0; i < 8; ++i) tot += sb[i];
  float mean = tot * (1.0f / Dd);
  __syncthreads();

  float s2 = 0.0f;
#pragma unroll
  for (int j = 0; j < 4; ++j) { float dd = v[j] - mean; s2 += dd * dd; }
#pragma unroll
  for (int d = 16; d >= 1; d >>= 1) s2 += __shfl_xor(s2, d, 32);
  if ((tid & 31) == 0) sb[tid >> 5] = s2;
  __syncthreads();
  float tot2 = 0.0f;
#pragma unroll
  for (int i = 0; i < 8; ++i) tot2 += sb[i];
  float rstd = rsqrtf(tot2 * (1.0f / Dd) + 1e-5f);

#pragma unroll
  for (int j = 0; j < 4; ++j) {
    int d = tid + j * 256;
    float ov = (v[j] - mean) * rstd * g[d] + bt[d];
    xout[(size_t)row * Dd + d] = ov;
    xh[(size_t)row * Dd + d]   = (half_t)ov;
  }
}

// ---------------------------------------------------------------------------
extern "C" void kernel_launch(void* const* d_in, const int* in_sizes, int n_in,
                              void* d_out, int out_size, void* d_ws, size_t ws_size,
                              hipStream_t stream) {
  const int*   tok  = (const int*)  d_in[0];
  const float* emb  = (const float*)d_in[1];
  const float* Wq   = (const float*)d_in[2];
  const float* bq   = (const float*)d_in[3];
  const float* Wk   = (const float*)d_in[4];
  const float* bk   = (const float*)d_in[5];
  const float* Wv   = (const float*)d_in[6];
  const float* bv   = (const float*)d_in[7];
  const float* Wo   = (const float*)d_in[8];
  const float* bo   = (const float*)d_in[9];
  const float* g1   = (const float*)d_in[10];
  const float* be1  = (const float*)d_in[11];
  const float* W1   = (const float*)d_in[12];
  const float* b1   = (const float*)d_in[13];
  const float* W2   = (const float*)d_in[14];
  const float* b2   = (const float*)d_in[15];
  const float* g2   = (const float*)d_in[16];
  const float* be2  = (const float*)d_in[17];

  char* p = (char*)d_ws;
  auto alloc = [&](size_t bytes) -> void* {
    void* r = (void*)p;
    p += (bytes + 255) & ~(size_t)255;
    return r;
  };
  const size_t DD = (size_t)Dd * Dd;       // 1M
  const size_t DF = (size_t)Dd * DFFc;     // 4M

  half_t* whQ = (half_t*)alloc(LLc * DD * 2);
  half_t* whK = (half_t*)alloc(LLc * DD * 2);
  half_t* whV = (half_t*)alloc(LLc * DD * 2);
  half_t* whO = (half_t*)alloc(LLc * DD * 2);
  half_t* wh1 = (half_t*)alloc(LLc * DF * 2);
  half_t* wh2 = (half_t*)alloc(LLc * DF * 2);
  float*  x   = (float*) alloc((size_t)MM * Dd * 4);
  float*  yb  = (float*) alloc((size_t)MM * Dd * 4);
  half_t* xh  = (half_t*)alloc((size_t)MM * Dd * 2);
  half_t* qh  = (half_t*)alloc((size_t)MM * Dd * 2);
  half_t* kh  = (half_t*)alloc((size_t)MM * Dd * 2);
  half_t* vh  = (half_t*)alloc((size_t)MM * Dd * 2);
  half_t* ah  = (half_t*)alloc((size_t)MM * Dd * 2);
  half_t* h1  = (half_t*)alloc((size_t)MM * DFFc * 2);

  // One-time (per launch) weight conversion into fragment-native f16 layout.
  for (int l = 0; l < LLc; ++l) {
    wconv_kernel<<<(unsigned)(DD / 256), 256, 0, stream>>>(Wq + l * DD, whQ + l * DD, Dd, Dd);
    wconv_kernel<<<(unsigned)(DD / 256), 256, 0, stream>>>(Wk + l * DD, whK + l * DD, Dd, Dd);
    wconv_kernel<<<(unsigned)(DD / 256), 256, 0, stream>>>(Wv + l * DD, whV + l * DD, Dd, Dd);
    wconv_kernel<<<(unsigned)(DD / 256), 256, 0, stream>>>(Wo + l * DD, whO + l * DD, Dd, Dd);
    wconv_kernel<<<(unsigned)(DF / 256), 256, 0, stream>>>(W1 + l * DF, wh1 + l * DF, Dd, DFFc);
    wconv_kernel<<<(unsigned)(DF / 256), 256, 0, stream>>>(W2 + l * DF, wh2 + l * DF, DFFc, Dd);
  }

  embed_kernel<<<MM, 256, 0, stream>>>(tok, emb, x, xh);

  dim3 gd(Dd / 64, MM / 128);       // N=1024 GEMMs (block tile 128x64)
  dim3 gf(DFFc / 64, MM / 128);     // N=4096 GEMM
  for (int l = 0; l < LLc; ++l) {
    gemm_f16<<<gd, 128, 0, stream>>>(xh, whQ + l * DD, bq + l * Dd, nullptr, qh, Dd, Dd, 0);
    gemm_f16<<<gd, 128, 0, stream>>>(xh, whK + l * DD, bk + l * Dd, nullptr, kh, Dd, Dd, 0);
    gemm_f16<<<gd, 128, 0, stream>>>(xh, whV + l * DD, bv + l * Dd, nullptr, vh, Dd, Dd, 0);
    attn_flash<<<dim3(Tt / 64, Hh, Bb), 128, 0, stream>>>(qh, kh, vh, ah);
    gemm_f16<<<gd, 128, 0, stream>>>(ah, whO + l * DD, bo + l * Dd, yb, nullptr, Dd, Dd, 0);
    ln_res<<<MM, 256, 0, stream>>>(x, yb, g1 + l * Dd, be1 + l * Dd, x, xh);
    gemm_f16<<<gf, 128, 0, stream>>>(xh, wh1 + l * DF, b1 + l * DFFc, nullptr, h1, DFFc, Dd, 1);
    gemm_f16<<<gd, 128, 0, stream>>>(h1, wh2 + l * DF, b2 + l * Dd, yb, nullptr, Dd, DFFc, 0);
    float* xo = (l == LLc - 1) ? (float*)d_out : x;
    ln_res<<<MM, 256, 0, stream>>>(x, yb, g2 + l * Dd, be2 + l * Dd, xo, xh);
  }
}
